// MyViT_32762010534327
// MI455X (gfx1250) — matrix operand order repaired and checked by reading, not compiled, not run on hardware
//
#include <hip/hip_runtime.h>

typedef __attribute__((ext_vector_type(16))) __bf16        v16bf;
typedef __attribute__((ext_vector_type(8)))  float         v8f;
typedef __attribute__((ext_vector_type(8)))  unsigned int  v8u;

#define BATCH  128
#define SEQ    197
#define NPAT   196
#define HID    768
#define K_DIM  768
#define IMG_C  3
#define IMG_H  224
#define IMG_W  224

#define WCVT_PLANE (HID * K_DIM)   // bf16 elements per plane (hi or lo)

// ---------------------------------------------------------------------------
// Pre-pass: split W (fp32) into bf16 hi/lo planes, packed 2 elements per dword.
// hi = truncate(x) (top 16 bits), lo = truncate(x - hi)  [x - hi is exact]
// ---------------------------------------------------------------------------
__global__ __launch_bounds__(256) void convert_w(
    const float* __restrict__ Wm,
    unsigned int* __restrict__ w_hi,    // WCVT_PLANE/2 dwords
    unsigned int* __restrict__ w_lo)
{
    const int idx = blockIdx.x * blockDim.x + threadIdx.x;  // pair index
    if (idx >= WCVT_PLANE / 2) return;
    const float2 f = reinterpret_cast<const float2*>(Wm)[idx];
    const unsigned int u0 = __float_as_uint(f.x);
    const unsigned int u1 = __float_as_uint(f.y);
    const float h0 = __uint_as_float(u0 & 0xffff0000u);
    const float h1 = __uint_as_float(u1 & 0xffff0000u);
    const unsigned int l0 = __float_as_uint(f.x - h0);
    const unsigned int l1 = __float_as_uint(f.y - h1);
    // pack {hi16(u1), hi16(u0)} -> one dword (little-endian elem order)
    w_hi[idx] = __builtin_amdgcn_perm(u1, u0, 0x07060302u);
    w_lo[idx] = __builtin_amdgcn_perm(l1, l0, 0x07060302u);
}

// ---------------------------------------------------------------------------
// Main GEMM: C[25088, 768] = patchify(images) @ W^T  (+bias +pos_embed)
// per wave: 16 M-rows x 64 N-cols, bf16x3 accumulation in f32 WMMA
// ---------------------------------------------------------------------------
__global__ __launch_bounds__(256) void vit_embed_gemm(
    const float* __restrict__ images,          // [128,3,224,224]
    const unsigned short* __restrict__ w_hi,   // [768,768] bf16 hi
    const unsigned short* __restrict__ w_lo,   // [768,768] bf16 lo
    const float* __restrict__ bias,            // [768]
    const float* __restrict__ pos,             // [197,768]
    float* __restrict__ out)                   // [128,197,768]
{
    const int lane = threadIdx.x & 31;
    const int wave = threadIdx.x >> 5;
    const int n0   = blockIdx.x * 64;                 // 12 column tiles
    const int m0   = blockIdx.y * 128 + wave * 16;    // 16 rows per wave

    // ---- A-side per-lane addressing (patchify gather) ----
    const int arow  = m0 + (lane & 15);
    const int bimg  = arow / NPAT;
    const int p     = arow - bimg * NPAT;
    const int pi    = p / 14;
    const int pj    = p - pi * 14;
    const long imgBase = (long)bimg * (IMG_C * IMG_H * IMG_W);
    const int  rowOff  = pi * 16;
    const int  colBase = pj * 16;

    const int bn      = lane & 15;
    const int kShiftA = (lane < 16) ? 0 : 16;
    const int kShiftB = (lane < 16) ? 0 : 16;

    v8f acc[4] = {v8f{}, v8f{}, v8f{}, v8f{}};

    #pragma unroll 2
    for (int k0 = 0; k0 < K_DIM; k0 += 32) {
        // ---- load A fragment (2 runs of 8 contiguous fp32), truncation split ----
        v8u ahw, alw;
        const int ka = k0 + kShiftA;
        #pragma unroll
        for (int run = 0; run < 2; ++run) {
            const int k = ka + run * 8;
            const int c = k >> 8;
            const int y = (k >> 4) & 15;
            const int x = k & 15;
            const float* aptr = images + imgBase +
                (long)(c * IMG_H + rowOff + y) * IMG_W + colBase + x;
            const float4 f0 = reinterpret_cast<const float4*>(aptr)[0];
            const float4 f1 = reinterpret_cast<const float4*>(aptr)[1];
            const float v[8] = {f0.x, f0.y, f0.z, f0.w, f1.x, f1.y, f1.z, f1.w};
            #pragma unroll
            for (int q = 0; q < 4; ++q) {
                const unsigned int u0 = __float_as_uint(v[2 * q]);
                const unsigned int u1 = __float_as_uint(v[2 * q + 1]);
                const float h0 = __uint_as_float(u0 & 0xffff0000u);
                const float h1 = __uint_as_float(u1 & 0xffff0000u);
                const unsigned int l0 = __float_as_uint(v[2 * q] - h0);
                const unsigned int l1 = __float_as_uint(v[2 * q + 1] - h1);
                ahw[run * 4 + q] = __builtin_amdgcn_perm(u1, u0, 0x07060302u);
                alw[run * 4 + q] = __builtin_amdgcn_perm(l1, l0, 0x07060302u);
            }
        }
        const v16bf a_hi = __builtin_bit_cast(v16bf, ahw);
        const v16bf a_lo = __builtin_bit_cast(v16bf, alw);

        // prefetch next A chunk (uniform condition; emits global_prefetch_b8)
        if (k0 + 32 < K_DIM) {
            const int kn = ka + 32;
            __builtin_prefetch(images + imgBase +
                (long)((kn >> 8) * IMG_H + rowOff + ((kn >> 4) & 15)) * IMG_W +
                colBase + (kn & 15), 0, 3);
        }

        // ---- four 16-column B tiles from pre-split bf16 planes ----
        #pragma unroll
        for (int j = 0; j < 4; ++j) {
            const long boff = (long)(n0 + j * 16 + bn) * K_DIM + k0 + kShiftB;
            const v16bf b_hi = *reinterpret_cast<const v16bf*>(w_hi + boff);
            const v16bf b_lo = *reinterpret_cast<const v16bf*>(w_lo + boff);
            acc[j] = __builtin_amdgcn_wmma_f32_16x16x32_bf16(
                false, a_hi, false, b_hi, (short)0, acc[j], false, false);
            acc[j] = __builtin_amdgcn_wmma_f32_16x16x32_bf16(
                false, a_hi, false, b_lo, (short)0, acc[j], false, false);
            acc[j] = __builtin_amdgcn_wmma_f32_16x16x32_bf16(
                false, a_lo, false, b_hi, (short)0, acc[j], false, false);
        }
    }

    // ---- epilogue: + bias + pos_embed, scatter into [b, p+1, h] ----
    const int col0 = n0 + bn;
    const int rOff = (lane < 16) ? 0 : 8;
    float bsv[4];
    #pragma unroll
    for (int j = 0; j < 4; ++j) bsv[j] = bias[col0 + j * 16];

    #pragma unroll
    for (int r = 0; r < 8; ++r) {
        const int mrow = m0 + r + rOff;
        const int bo   = mrow / NPAT;
        const int po   = mrow - bo * NPAT;
        const long ob  = ((long)bo * SEQ + po + 1) * HID;
        const long pb  = (long)(po + 1) * HID;
        #pragma unroll
        for (int j = 0; j < 4; ++j) {
            const int col = col0 + j * 16;
            out[ob + col] = acc[j][r] + bsv[j] + pos[pb + col];
        }
    }
}

// class-token row: out[b, 0, h] = class_token[h] + pos_embed[0, h]
__global__ __launch_bounds__(256) void vit_cls_row(
    const float* __restrict__ ct, const float* __restrict__ pos,
    float* __restrict__ out)
{
    const int idx = blockIdx.x * blockDim.x + threadIdx.x;
    if (idx >= BATCH * HID) return;
    const int bo = idx / HID;
    const int h  = idx - bo * HID;
    out[(long)bo * SEQ * HID + h] = ct[h] + pos[h];
}

extern "C" void kernel_launch(void* const* d_in, const int* in_sizes, int n_in,
                              void* d_out, int out_size, void* d_ws, size_t ws_size,
                              hipStream_t stream) {
    const float* images = (const float*)d_in[0];   // [128,3,224,224]
    const float* Wm     = (const float*)d_in[1];   // [768,768]
    const float* bias   = (const float*)d_in[2];   // [768]
    const float* ct     = (const float*)d_in[3];   // [1,768]
    const float* pos    = (const float*)d_in[4];   // [197,768]
    float* out = (float*)d_out;

    unsigned short* w_hi = (unsigned short*)d_ws;            // 1.125 MB
    unsigned short* w_lo = w_hi + WCVT_PLANE;                // 1.125 MB

    // 1) split W into bf16 hi/lo planes (once per launch, same stream)
    convert_w<<<(WCVT_PLANE / 2 + 255) / 256, 256, 0, stream>>>(
        Wm, (unsigned int*)w_hi, (unsigned int*)w_lo);

    // 2) main WMMA GEMM with fused patchify + bias + pos_embed
    dim3 grid(HID / 64, (BATCH * NPAT) / 128);     // (12, 196)
    vit_embed_gemm<<<grid, 256, 0, stream>>>(images, w_hi, w_lo, bias, pos, out);

    // 3) class-token rows
    vit_cls_row<<<(BATCH * HID + 255) / 256, 256, 0, stream>>>(ct, pos, out);
}
